// NoOnlineLogitNormalizationLayer_12111807775040
// MI455X (gfx1250) — matrix operand order repaired and checked by reading, not compiled, not run on hardware
//
#include <hip/hip_runtime.h>
#include <stdint.h>

// ---------------- problem dims ----------------
#define B_   4
#define N_   4096
#define D_   1024
#define H_   16
#define DH_  64
#define ROWS (B_ * N_)   // 16384
#define NCAT (3 * D_)    // 3072 (Q|K|V concatenated output features)

// stage-1 GEMM tiling
#define BM   128
#define BN   128
#define BK   32
#define LDSL 40          // padded LDS row stride (elements): 20-bank rotation/row

// ---------------- vector types ----------------
typedef __attribute__((ext_vector_type(16))) __bf16         bf16x16;
typedef __attribute__((ext_vector_type(8)))  float          f32x8;
typedef __attribute__((ext_vector_type(4)))  unsigned int   u32x4;
typedef __attribute__((ext_vector_type(4)))  int            i32x4;
typedef __attribute__((ext_vector_type(4)))  float          f32x4;
typedef __attribute__((ext_vector_type(4)))  unsigned short u16x4;

union FragAB {
    bf16x16 v;
    u32x4   u[2];
};

#if defined(__has_builtin)
#if __has_builtin(__builtin_amdgcn_global_load_async_to_lds_b128) && \
    __has_builtin(__builtin_amdgcn_s_wait_asynccnt)
#define ASYNC_LDS 1
#endif
#endif

#ifdef ASYNC_LDS
typedef i32x4 __attribute__((address_space(1)))* g_i32x4p;   // global
typedef i32x4 __attribute__((address_space(3)))* l_i32x4p;   // LDS
#endif

// 16-byte global -> LDS copy (async-to-LDS when available)
static __device__ inline void cp16_to_lds(const unsigned short* g, unsigned short* l) {
#ifdef ASYNC_LDS
    __builtin_amdgcn_global_load_async_to_lds_b128((g_i32x4p)(g), (l_i32x4p)(l), 0, 0);
#else
    *(u32x4*)l = *(const u32x4*)g;
#endif
}

static __device__ inline void lds_fill_wait_barrier() {
#ifdef ASYNC_LDS
    __builtin_amdgcn_s_wait_asynccnt(0);
#endif
    __syncthreads();
}

// round-to-nearest-even f32 -> bf16
static __device__ inline unsigned short f2bf(float f) {
    unsigned int u = __float_as_uint(f);
    unsigned int r = (u + 0x7FFFu + ((u >> 16) & 1u)) >> 16;
    return (unsigned short)r;
}

// A-matrix 16x32 bf16 fragment (ISA 7.12.2): lane m = l&15 holds row m.
// lanes 0-15: K = 0..7 (v0-3), 16..23 (v4-7); lanes 16-31: K = 8..15, 24..31.
static __device__ inline bf16x16 load_a(const unsigned short* base, int ld, int lane) {
    int m  = lane & 15;
    int kh = lane >> 4;
    const unsigned short* row = base + (size_t)m * ld;
    FragAB f;
    f.u[0] = *(const u32x4*)(row + kh * 8);
    f.u[1] = *(const u32x4*)(row + 16 + kh * 8);
    return f.v;
}

// B-matrix 32x16 bf16 fragment: lane holds column n = l&15; lanes 0-15 K=0..7,16..23,
// lanes 16-31 K=8..15,24..31. Row n of `base` holds B[:,n] contiguous along K.
static __device__ inline bf16x16 load_b(const unsigned short* base, int ld, int lane) {
    int n  = lane & 15;
    int kh = lane >> 4;
    const unsigned short* row = base + (size_t)n * ld;
    FragAB f;
    f.u[0] = *(const u32x4*)(row + kh * 8);
    f.u[1] = *(const u32x4*)(row + 16 + kh * 8);
    return f.v;
}

// ---------------- conversion kernels ----------------
__global__ void cvt_f32_bf16(const float* __restrict__ in,
                             unsigned short* __restrict__ out, int n4) {
    int i = blockIdx.x * blockDim.x + threadIdx.x;
    if (i < n4) {
        f32x4 f = ((const f32x4*)in)[i];
        u16x4 o;
        o[0] = f2bf(f[0]); o[1] = f2bf(f[1]); o[2] = f2bf(f[2]); o[3] = f2bf(f[3]);
        ((u16x4*)out)[i] = o;
    }
}

__global__ void bias_concat(const float* __restrict__ bq, const float* __restrict__ bk,
                            const float* __restrict__ bv, float* __restrict__ bcat) {
    int i = blockIdx.x * blockDim.x + threadIdx.x;
    if (i < D_) {
        bcat[i]          = bq[i];
        bcat[D_ + i]     = bk[i];
        bcat[2 * D_ + i] = bv[i];
    }
}

// ---------------- stage 1: fused QKV projection GEMM ----------------
// C[16384 x 3072] = X[16384 x 1024] * Wcat^T + bias.
// Block: 128x128 tile, LDS double-buffered 32-wide K slices (async fill).
// Wave: 64x32 register tile (4x2 WMMA accumulators).
// Q stored row-major [b,n,h,dh]; K,V stored transposed per head [b,h,dh,n].
__global__ void __launch_bounds__(256)
qkv_gemm(const unsigned short* __restrict__ x,
         const unsigned short* __restrict__ w,
         const float* __restrict__ bcat,
         unsigned short* __restrict__ q,
         unsigned short* __restrict__ kT,
         unsigned short* __restrict__ vT) {
    __shared__ unsigned short ldsA[2][BM * LDSL];   // 2 * 10 KB
    __shared__ unsigned short ldsB[2][BN * LDSL];   // 2 * 10 KB

    int tid  = threadIdx.x;
    int lane = tid & 31;
    int wave = tid >> 5;
    int wm = wave & 1;            // 2 waves along M (64 rows each)
    int wn = wave >> 1;           // 4 waves along N (32 cols each)

    const unsigned short* Ag = x + (size_t)blockIdx.x * BM * D_;
    const unsigned short* Bg = w + (size_t)blockIdx.y * BN * D_;

    // issue one 32-wide K slice into LDS buffer `buf`
    auto issue = [&](int ks, int buf) {
        int kk = ks * BK;
        // A tile: 128 rows x 64 B = 512 x 16 B chunks; 2 per thread
        int c0 = tid * 2;
        for (int c = c0; c < c0 + 2; ++c) {
            int row = c >> 2, col = (c & 3) * 8;
            cp16_to_lds(Ag + (size_t)row * D_ + kk + col,
                        &ldsA[buf][row * LDSL + col]);
        }
        // B tile: 128 feature-rows x 64 B
        for (int c = c0; c < c0 + 2; ++c) {
            int row = c >> 2, col = (c & 3) * 8;
            cp16_to_lds(Bg + (size_t)row * D_ + kk + col,
                        &ldsB[buf][row * LDSL + col]);
        }
    };

    f32x8 acc[4][2] = {};

    issue(0, 0);
    const int KSTEPS = D_ / BK;   // 32
    for (int ks = 0; ks < KSTEPS; ++ks) {
        int buf = ks & 1;
        lds_fill_wait_barrier();              // slice `ks` resident, all waves synced
        if (ks + 1 < KSTEPS) issue(ks + 1, buf ^ 1);

        const unsigned short* As = ldsA[buf];
        const unsigned short* Bs = ldsB[buf];
        bf16x16 af[4], bfr[2];
        for (int mt = 0; mt < 4; ++mt)
            af[mt] = load_a(As + (wm * 64 + mt * 16) * LDSL, LDSL, lane);
        for (int nt = 0; nt < 2; ++nt)
            bfr[nt] = load_b(Bs + (wn * 32 + nt * 16) * LDSL, LDSL, lane);
        for (int mt = 0; mt < 4; ++mt)
            for (int nt = 0; nt < 2; ++nt)
                acc[mt][nt] = __builtin_amdgcn_wmma_f32_16x16x32_bf16(
                    false, af[mt], false, bfr[nt], (short)0, acc[mt][nt], false, false);
    }

    // epilogue: bias add, bf16 convert, scatter Q / K^T / V^T
    for (int nt = 0; nt < 2; ++nt) {
        int c  = blockIdx.y * BN + wn * 32 + nt * 16 + (lane & 15);
        float bi = bcat[c];
        for (int mt = 0; mt < 4; ++mt) {
            int rbase = blockIdx.x * BM + wm * 64 + mt * 16 + 8 * (lane >> 4);
            if (c < D_) {                          // Q row-major
                for (int j = 0; j < 8; ++j)
                    q[(size_t)(rbase + j) * D_ + c] = f2bf(acc[mt][nt][j] + bi);
            } else if (c < 2 * D_) {               // K -> [b,h,dh,n]
                int d = c - D_;
                int h = d >> 6, dd = d & 63;
                for (int j = 0; j < 8; ++j) {
                    int r = rbase + j, bb = r >> 12, n = r & (N_ - 1);
                    kT[(((size_t)(bb * H_ + h)) * DH_ + dd) * N_ + n] =
                        f2bf(acc[mt][nt][j] + bi);
                }
            } else {                               // V -> [b,h,dh,n]
                int d = c - 2 * D_;
                int h = d >> 6, dd = d & 63;
                for (int j = 0; j < 8; ++j) {
                    int r = rbase + j, bb = r >> 12, n = r & (N_ - 1);
                    vT[(((size_t)(bb * H_ + h)) * DH_ + dd) * N_ + n] =
                        f2bf(acc[mt][nt][j] + bi);
                }
            }
        }
    }
}

// ---------------- stage 2: ktv[d,e] = sum_n k[n,d] v[n,e] per (b,h) ----------------
__global__ void ktv_gemm(const unsigned short* __restrict__ kT,
                         const unsigned short* __restrict__ vT,
                         unsigned short* __restrict__ ktvT) {
    int lane = threadIdx.x & 31;
    int wave = threadIdx.x >> 5;
    int wid  = blockIdx.x * 8 + wave;    // 0..1023
    int bh   = wid >> 4;                 // 0..63
    int tile = wid & 15;
    int dt = tile >> 2, et = tile & 3;

    const unsigned short* abase = kT + (size_t)bh * DH_ * N_ + (size_t)dt * 16 * N_;
    const unsigned short* bbase = vT + (size_t)bh * DH_ * N_ + (size_t)et * 16 * N_;

    f32x8 acc = {};
    for (int kk = 0; kk < N_; kk += 32) {
        bf16x16 a = load_a(abase + kk, N_, lane);
        bf16x16 b = load_b(bbase + kk, N_, lane);
        acc = __builtin_amdgcn_wmma_f32_16x16x32_bf16(false, a, false, b,
                                                      (short)0, acc, false, false);
    }

    int e     = et * 16 + (lane & 15);
    int dbase = dt * 16 + 8 * (lane >> 4);
    unsigned short* dst = ktvT + (size_t)bh * DH_ * DH_ + (size_t)e * DH_ + dbase;
    for (int j = 0; j < 8; ++j) dst[j] = f2bf(acc[j]);
}

// ---------------- stage 3: out[n, h*64+e] = scale * sum_d q[n,h,d] ktv[d,e] ----------------
__global__ void out_gemm(const unsigned short* __restrict__ q,
                         const unsigned short* __restrict__ ktvT,
                         float* __restrict__ out, float scale) {
    int lane = threadIdx.x & 31;
    int wave = threadIdx.x >> 5;
    int bh    = blockIdx.y;              // 0..63
    int et    = wave & 3;                // 0..3
    int ntile = blockIdx.x * 2 + (wave >> 2);  // 0..255
    int bb = bh >> 4, h = bh & 15;

    const unsigned short* abase = q + ((size_t)bb * N_ + (size_t)ntile * 16) * D_ + h * DH_;
    const unsigned short* bbase = ktvT + (size_t)bh * DH_ * DH_ + (size_t)et * 16 * DH_;

    f32x8 acc = {};
    for (int kk = 0; kk < DH_; kk += 32) {
        bf16x16 a = load_a(abase + kk, D_, lane);
        bf16x16 b = load_b(bbase + kk, DH_, lane);
        acc = __builtin_amdgcn_wmma_f32_16x16x32_bf16(false, a, false, b,
                                                      (short)0, acc, false, false);
    }

    int e = h * DH_ + et * 16 + (lane & 15);
    for (int j = 0; j < 8; ++j) {
        int n = ntile * 16 + j + 8 * (lane >> 4);
        out[((size_t)bb * N_ + n) * D_ + e] = acc[j] * scale;
    }
}

// ---------------- launcher ----------------
extern "C" void kernel_launch(void* const* d_in, const int* in_sizes, int n_in,
                              void* d_out, int out_size, void* d_ws, size_t ws_size,
                              hipStream_t stream) {
    (void)in_sizes; (void)n_in; (void)out_size; (void)ws_size;

    const float* tens = (const float*)d_in[0];
    const float* Wq   = (const float*)d_in[1];
    const float* bq   = (const float*)d_in[2];
    const float* Wk   = (const float*)d_in[3];
    const float* bk   = (const float*)d_in[4];
    const float* Wv   = (const float*)d_in[5];
    const float* bv   = (const float*)d_in[6];
    float* out = (float*)d_out;

    // workspace carve-up (all offsets 256B-aligned)
    char* ws = (char*)d_ws;
    size_t o = 0;
    unsigned short* x_bf = (unsigned short*)(ws + o); o += (size_t)ROWS * D_ * 2;   // 32 MB
    unsigned short* w_bf = (unsigned short*)(ws + o); o += (size_t)NCAT * D_ * 2;   //  6 MB
    float*          bcat = (float*)(ws + o);          o += (size_t)NCAT * 4;        // 12 KB
    unsigned short* q_bf = (unsigned short*)(ws + o); o += (size_t)ROWS * D_ * 2;   // 32 MB
    unsigned short* kT   = (unsigned short*)(ws + o); o += (size_t)B_ * H_ * DH_ * N_ * 2; // 32 MB
    unsigned short* vT   = (unsigned short*)(ws + o); o += (size_t)B_ * H_ * DH_ * N_ * 2; // 32 MB
    unsigned short* ktvT = (unsigned short*)(ws + o); o += (size_t)B_ * H_ * DH_ * DH_ * 2; // 0.5 MB

    // stage 0: precision conversion (bf16 operands fit in 192 MB L2)
    int n4x = ROWS * D_ / 4;
    cvt_f32_bf16<<<(n4x + 255) / 256, 256, 0, stream>>>(tens, x_bf, n4x);
    int n4w = D_ * D_ / 4;
    cvt_f32_bf16<<<(n4w + 255) / 256, 256, 0, stream>>>(Wq, w_bf,               n4w);
    cvt_f32_bf16<<<(n4w + 255) / 256, 256, 0, stream>>>(Wk, w_bf + D_ * D_,     n4w);
    cvt_f32_bf16<<<(n4w + 255) / 256, 256, 0, stream>>>(Wv, w_bf + 2 * D_ * D_, n4w);
    bias_concat<<<4, 256, 0, stream>>>(bq, bk, bv, bcat);

    // stage 1: fused QKV projection (M=16384, N=3072, K=1024), 128x128 blocks
    qkv_gemm<<<dim3(ROWS / BM, NCAT / BN), 256, 0, stream>>>(x_bf, w_bf, bcat,
                                                             q_bf, kT, vT);

    // stage 2: per-(b,h) K^T V, 64x64 over N=4096
    ktv_gemm<<<128, 256, 0, stream>>>(kT, vT, ktvT);

    // stage 3: out = scale * Q @ ktv ; scale = (1/sqrt(64))/sqrt(4096) = 1/512
    out_gemm<<<dim3(N_ / 32, B_ * H_), 256, 0, stream>>>(q_bf, ktvT, out,
                                                         1.0f / 512.0f);
}
